// SAGE_68247030333463
// MI455X (gfx1250) — matrix operand order repaired and checked by reading, not compiled, not run on hardware
//
#include <hip/hip_runtime.h>
#include <hip/hip_bf16.h>
#include <math.h>

// ---------------- types ----------------
typedef _Float16 half8  __attribute__((ext_vector_type(8)));
typedef _Float16 half16 __attribute__((ext_vector_type(16)));
typedef float    floatx8 __attribute__((ext_vector_type(8)));

// Problem constants (match reference)
#define N0c 1000000
#define N1c 102400
#define N2c 10240
#define N3c 1024
#define D_INc 128
#define D_Hc 256
#define D_OUTc 40

// ---------------- kernels ----------------

// Edge scatter-add: sum[dst[e]][d] += h[src[e]][d]; cnt[dst[e]] += 1 (once per edge).
// One thread per (edge, feature). D is a power of two (128 or 256).
__global__ void k_scatter_add(const float* __restrict__ h,
                              const int* __restrict__ src,
                              const int* __restrict__ dst,
                              float* __restrict__ sum,
                              float* __restrict__ cnt,
                              int E, int logD) {
    long long i = (long long)blockIdx.x * blockDim.x + threadIdx.x;
    int D = 1 << logD;
    long long total = (long long)E << logD;
    if (i >= total) return;
    int e = (int)(i >> logD);
    int d = (int)(i & (D - 1));
    int s = src[e];
    int t = dst[e];
    atomicAdd(&sum[(long long)t * D + d], h[(long long)s * D + d]);
    if (d == 0) atomicAdd(&cnt[t], 1.0f);
}

// Build fused f16 A matrix: A[m][0..D-1] = sum[m]/max(cnt,1), A[m][D..2D-1] = h_tgt[m].
__global__ void k_build_A(const float* __restrict__ sum,
                          const float* __restrict__ cnt,
                          const float* __restrict__ h_tgt,
                          _Float16* __restrict__ A,
                          int n_tgt, int logD) {
    long long i = (long long)blockIdx.x * blockDim.x + threadIdx.x;
    int D = 1 << logD;
    long long total = (long long)n_tgt << (logD + 1);
    if (i >= total) return;
    int m  = (int)(i >> (logD + 1));
    int d2 = (int)(i & (2 * D - 1));
    float v;
    if (d2 < D) {
        float c = cnt[m];
        v = sum[(long long)m * D + d2] / fmaxf(c, 1.0f);
    } else {
        v = h_tgt[(long long)m * D + (d2 - D)];
    }
    A[i] = (_Float16)v;
}

// Pack stacked weights B = [Wl ; Wr] (K'=2*Kh rows, N cols, f32 row-major) into the
// v_wmma_f32_16x16x32_f16 B-operand VGPR layout, tile-contiguous:
//   tile (nt, kt) occupies 512 halves; within a tile, lane L (0..31) holds 16 halves:
//   lane<16  -> column n=nt*16+L,      K = kt*32 + {0..7, 16..23} packed 2/VGPR
//   lane>=16 -> column n=nt*16+(L-16), K = kt*32 + {8..15, 24..31}
// Each GEMM lane then loads its operand with ONE contiguous 32-byte load.
__global__ void k_pack_B(const float* __restrict__ Wl,
                         const float* __restrict__ Wr,
                         _Float16* __restrict__ Bp,
                         int Kh, int Nreal, int KT, int Ntiles) {
    int i = blockIdx.x * blockDim.x + threadIdx.x;
    int total = KT * Ntiles * 512;
    if (i >= total) return;
    int tile = i >> 9;
    int off  = i & 511;
    int lane = off >> 4;
    int q    = off & 15;
    int j    = q >> 1;
    int slot = q & 1;
    int nt   = tile / KT;
    int kt   = tile - nt * KT;
    int kk   = ((q >> 3) << 4) + ((lane >> 4) << 3) + (q & 7); (void)j; (void)slot;   // 0..31 within tile
    int k    = kt * 32 + kk;
    int n    = nt * 16 + (lane & 15);
    float v = 0.0f;
    if (n < Nreal) {
        v = (k < Kh) ? Wl[k * Nreal + n] : Wr[(k - Kh) * Nreal + n];
    }
    Bp[i] = (_Float16)v;
}

// WMMA GEMM: out[M x Nreal] = A[M x K'] * Bpacked + bias, optional ReLU.
// One wave32 per 16x16 output tile; K' = KT*32.
__global__ void k_gemm_wmma(const _Float16* __restrict__ A,
                            const _Float16* __restrict__ Bp,
                            const float* __restrict__ bias,
                            float* __restrict__ out,
                            int Mtiles, int Ntiles, int KT, int Nreal, int act) {
    int gtid = blockIdx.x * blockDim.x + threadIdx.x;
    int wid  = gtid >> 5;          // wave id (uniform per wave)
    int lane = gtid & 31;
    if (wid >= Mtiles * Ntiles) return;  // wave-uniform: EXEC stays all-1s for WMMA
    int mt = wid / Ntiles;
    int nt = wid - mt * Ntiles;
    int ln = lane & 15;
    int hi = lane >> 4;
    int K  = KT * 32;

    // A operand: row (mt*16+ln); lane<16 takes K {k0..k0+7, k0+16..k0+23},
    // lane>=16 takes K {k0+8..k0+15, k0+24..k0+31}  (ISA 16-bit A layout).
    const _Float16* arow  = A  + (long long)(mt * 16 + ln) * K + hi * 8;
    const _Float16* btile = Bp + (long long)nt * KT * 512 + lane * 16;

    floatx8 acc = {0.f, 0.f, 0.f, 0.f, 0.f, 0.f, 0.f, 0.f};
    for (int kt = 0; kt < KT; ++kt) {
        half8 alo = *(const half8*)(arow);
        half8 ahi = *(const half8*)(arow + 16);
        half16 a = __builtin_shufflevector(alo, ahi,
                      0,1,2,3,4,5,6,7,8,9,10,11,12,13,14,15);
        half16 b = *(const half16*)(btile);
        acc = __builtin_amdgcn_wmma_f32_16x16x32_f16(
                  false, a, false, b, (short)0, acc, false, false);
        arow  += 32;
        btile += 512;
    }

    int n = nt * 16 + ln;
    if (n < Nreal) {
        float bi = bias[n];
        #pragma unroll
        for (int i = 0; i < 8; ++i) {
            int m = mt * 16 + hi * 8 + i;   // C/D layout: VGPR i -> row 8*hi+i
            float v = acc[i] + bi;
            if (act) v = fmaxf(v, 0.0f);
            out[(long long)m * Nreal + n] = v;
        }
    }
}

// In-place log-softmax over rows of C columns (C <= 64); one wave32 per row.
__global__ void k_log_softmax(float* __restrict__ io, int C) {
    int r = blockIdx.x;
    int lane = threadIdx.x;
    float v0 = (lane < C)      ? io[r * C + lane]      : -INFINITY;
    float v1 = (lane + 32 < C) ? io[r * C + lane + 32] : -INFINITY;
    float m = fmaxf(v0, v1);
    #pragma unroll
    for (int o = 16; o > 0; o >>= 1) m = fmaxf(m, __shfl_xor(m, o, 32));
    float s = 0.0f;
    if (lane < C)      s += expf(v0 - m);
    if (lane + 32 < C) s += expf(v1 - m);
    #pragma unroll
    for (int o = 16; o > 0; o >>= 1) s += __shfl_xor(s, o, 32);
    float ls = logf(s);
    if (lane < C)      io[r * C + lane]      = v0 - m - ls;
    if (lane + 32 < C) io[r * C + lane + 32] = v1 - m - ls;
}

// ---------------- host orchestration ----------------
static inline unsigned int nblk(long long total, int bs) {
    return (unsigned int)((total + bs - 1) / bs);
}

extern "C" void kernel_launch(void* const* d_in, const int* in_sizes, int n_in,
                              void* d_out, int out_size, void* d_ws, size_t ws_size,
                              hipStream_t stream) {
    (void)in_sizes; (void)n_in; (void)out_size; (void)ws_size;

    const float* x    = (const float*)d_in[0];
    const int*   src0 = (const int*)d_in[1];
    const int*   dst0 = (const int*)d_in[2];
    const int*   src1 = (const int*)d_in[3];
    const int*   dst1 = (const int*)d_in[4];
    const int*   src2 = (const int*)d_in[5];
    const int*   dst2 = (const int*)d_in[6];
    const float* Wl0  = (const float*)d_in[7];
    const float* bl0  = (const float*)d_in[8];
    const float* Wr0  = (const float*)d_in[9];
    const float* Wl1  = (const float*)d_in[10];
    const float* bl1  = (const float*)d_in[11];
    const float* Wr1  = (const float*)d_in[12];
    const float* Wl2  = (const float*)d_in[13];
    const float* bl2  = (const float*)d_in[14];
    const float* Wr2  = (const float*)d_in[15];

    const int E0 = in_sizes[1];  // 1,024,000
    const int E1 = in_sizes[3];  //   102,400
    const int E2 = in_sizes[5];  //    10,240

    // ---- workspace carve (256B aligned) ----
    char* w = (char*)d_ws;
    auto carve = [&](size_t bytes) -> void* {
        void* p = (void*)w;
        w += (bytes + 255) & ~(size_t)255;
        return p;
    };
    float*     sumb = (float*)    carve((size_t)N1c * D_INc * sizeof(float)); // 52.4MB (max of all layers)
    float*     cntb = (float*)    carve((size_t)N1c * sizeof(float));
    _Float16*  Ab   = (_Float16*) carve((size_t)N1c * 2 * D_INc * sizeof(_Float16)); // 52.4MB (max)
    float*     h1   = (float*)    carve((size_t)N1c * D_Hc * sizeof(float));  // 104.8MB
    float*     h2   = (float*)    carve((size_t)N2c * D_Hc * sizeof(float));  // 10.5MB
    _Float16*  B0p  = (_Float16*) carve((size_t)256 * 256 * sizeof(_Float16));
    _Float16*  B1p  = (_Float16*) carve((size_t)512 * 256 * sizeof(_Float16));
    _Float16*  B2p  = (_Float16*) carve((size_t)512 * 48  * sizeof(_Float16));

    const int BS = 256;

    // ---- pack weights into WMMA B-operand layout ----
    // Layer0: K'=256 (KT=8),  N=256 -> Ntiles=16
    k_pack_B<<<nblk(8 * 16 * 512, BS), BS, 0, stream>>>(Wl0, Wr0, B0p, 128, 256, 8, 16);
    // Layer1: K'=512 (KT=16), N=256 -> Ntiles=16
    k_pack_B<<<nblk(16 * 16 * 512, BS), BS, 0, stream>>>(Wl1, Wr1, B1p, 256, 256, 16, 16);
    // Layer2: K'=512 (KT=16), N=40 padded to 48 -> Ntiles=3
    k_pack_B<<<nblk(16 * 3 * 512, BS), BS, 0, stream>>>(Wl2, Wr2, B2p, 256, 40, 16, 3);

    // ================= Layer 0 =================
    hipMemsetAsync(sumb, 0, (size_t)N1c * D_INc * sizeof(float), stream);
    hipMemsetAsync(cntb, 0, (size_t)N1c * sizeof(float), stream);
    k_scatter_add<<<nblk((long long)E0 << 7, BS), BS, 0, stream>>>(
        x, src0, dst0, sumb, cntb, E0, 7);
    k_build_A<<<nblk((long long)N1c << 8, BS), BS, 0, stream>>>(
        sumb, cntb, x, Ab, N1c, 7);
    // M=102400 (6400 tiles), N=256 (16 tiles), K'=256
    k_gemm_wmma<<<nblk((long long)(6400 * 16) * 32, BS), BS, 0, stream>>>(
        Ab, B0p, bl0, h1, 6400, 16, 8, 256, 1);

    // ================= Layer 1 =================
    hipMemsetAsync(sumb, 0, (size_t)N2c * D_Hc * sizeof(float), stream);
    hipMemsetAsync(cntb, 0, (size_t)N2c * sizeof(float), stream);
    k_scatter_add<<<nblk((long long)E1 << 8, BS), BS, 0, stream>>>(
        h1, src1, dst1, sumb, cntb, E1, 8);
    k_build_A<<<nblk((long long)N2c << 9, BS), BS, 0, stream>>>(
        sumb, cntb, h1, Ab, N2c, 8);
    // M=10240 (640 tiles), N=256 (16 tiles), K'=512
    k_gemm_wmma<<<nblk((long long)(640 * 16) * 32, BS), BS, 0, stream>>>(
        Ab, B1p, bl1, h2, 640, 16, 16, 256, 1);

    // ================= Layer 2 =================
    hipMemsetAsync(sumb, 0, (size_t)N3c * D_Hc * sizeof(float), stream);
    hipMemsetAsync(cntb, 0, (size_t)N3c * sizeof(float), stream);
    k_scatter_add<<<nblk((long long)E2 << 8, BS), BS, 0, stream>>>(
        h2, src2, dst2, sumb, cntb, E2, 8);
    k_build_A<<<nblk((long long)N3c << 9, BS), BS, 0, stream>>>(
        sumb, cntb, h2, Ab, N3c, 8);
    // M=1024 (64 tiles), Npad=48 (3 tiles), K'=512, write valid 40 cols to d_out
    float* outf = (float*)d_out;
    k_gemm_wmma<<<nblk((long long)(64 * 3) * 32, BS), BS, 0, stream>>>(
        Ab, B2p, bl2, outf, 64, 3, 16, 40, 0);

    // log-softmax in place on d_out (1024 rows x 40)
    k_log_softmax<<<N3c, 32, 0, stream>>>(outf, 40);
}
